// AngularLoss_17884243820950
// MI455X (gfx1250) — matrix operand order repaired and checked by reading, not compiled, not run on hardware
//
#include <hip/hip_runtime.h>
#include <hip/hip_bf16.h>
#include <math.h>

typedef _Float16 v16h __attribute__((ext_vector_type(16)));
typedef float    v8f  __attribute__((ext_vector_type(8)));
typedef float    v4f  __attribute__((ext_vector_type(4)));

#define B_SIZE 512
#define D_SIZE 512
#define ALPHA_RAD 0.7853981633974483f   /* radians(45) */

// ---------------------------------------------------------------------------
// Kernel 0: zero the loss/count accumulators (graph-safe, no memset).
// ---------------------------------------------------------------------------
__global__ void ang_init_kernel(float* loss_sum, unsigned int* count) {
    if (threadIdx.x == 0) { *loss_sum = 0.0f; *count = 0u; }
}

// ---------------------------------------------------------------------------
// Kernel 1: L2-normalize each row, emit f16 matrix for the WMMA GEMM.
// One block (256 threads) per row of 512 floats.
// ---------------------------------------------------------------------------
__global__ void ang_normalize_kernel(const float* __restrict__ X,
                                     _Float16* __restrict__ Nh) {
    __shared__ float red[256];
    const int row = blockIdx.x;
    const int t   = threadIdx.x;
    const float x0 = X[row * D_SIZE + t];
    const float x1 = X[row * D_SIZE + t + 256];
    red[t] = x0 * x0 + x1 * x1;
    __syncthreads();
    for (int s = 128; s > 0; s >>= 1) {
        if (t < s) red[t] += red[t + s];
        __syncthreads();
    }
    const float scale = 1.0f / fmaxf(sqrtf(red[0]), 1e-12f);
    Nh[row * D_SIZE + t]       = (_Float16)(x0 * scale);
    Nh[row * D_SIZE + t + 256] = (_Float16)(x1 * scale);
}

// ---------------------------------------------------------------------------
// Kernel 2: sim = N * N^T via v_wmma_f32_16x16x32_f16, then ang = acos(clamp).
// One wave (32 lanes) owns one 16x16 output tile; 8 waves per block.
// Because C = N*N^T, the B fragment's column n is just row (colBase+n) of N,
// so both fragments load contiguous 16-byte chunks of N per the ISA layouts:
//   A 16x32 : lane(l,h) holds row l, K in {h*8..h*8+7, 16+h*8..16+h*8+7}
//   B 32x16 : lane(l,h) holds col l, K in {h*8..h*8+7, 16+h*8..16+h*8+7}
// ---------------------------------------------------------------------------
__global__ void ang_wmma_sim_kernel(const _Float16* __restrict__ Nh,
                                    float* __restrict__ ang) {
    const int lane = threadIdx.x & 31;
    const int wave = threadIdx.x >> 5;
    const int tile = blockIdx.x * 8 + wave;
    const int tileRow = tile >> 5;          // 32 tiles along each 512 dim
    const int tileCol = tile & 31;
    const int rowBase = tileRow * 16;
    const int colBase = tileCol * 16;
    const int h = lane >> 4;                // half-wave select
    const int l = lane & 15;                // row (A) / col (B) index

    union Frag { v16h v; v4f f[2]; };

    v8f c = {};
    for (int kBase = 0; kBase < D_SIZE; kBase += 32) {
        Frag a, b;
        // A: row rowBase+l, two 8-half chunks at K = kBase+h*8 and +16
        const v4f* pa = (const v4f*)(Nh + (rowBase + l) * D_SIZE + kBase + h * 8);
        a.f[0] = pa[0];
        a.f[1] = pa[2];                     // +16 halves = +2 v4f
        // B: col colBase+l = row colBase+l of N, two 8-half chunks at K = kBase+h*8 and +16
        const v4f* pb = (const v4f*)(Nh + (colBase + l) * D_SIZE + kBase + h * 8);
        b.f[0] = pb[0];
        b.f[1] = pb[2];
        c = __builtin_amdgcn_wmma_f32_16x16x32_f16(
                /*neg_a=*/false, a.v, /*neg_b=*/false, b.v,
                /*c_mod=*/(short)0, c, /*reuse_a=*/false, /*reuse_b=*/false);
    }

    // C/D layout: VGPR r holds M = r + 8*h, N = l
    #pragma unroll
    for (int r = 0; r < 8; ++r) {
        float s = c[r];
        s = fminf(1.0f, fmaxf(-1.0f, s));
        const int m = r + h * 8;
        ang[(rowBase + m) * B_SIZE + colBase + l] = acosf(s);
    }
}

// ---------------------------------------------------------------------------
// Kernel 3: per-anchor masked pairwise reduction.
// One block per anchor i. Row of angles + pos/neg flags cached in LDS.
// loss_i = sum_{p in P_i} sum_{n in N_i} relu(ang[i,p] + alpha - ang[i,n])
// count_i = |P_i| * |N_i|
// ---------------------------------------------------------------------------
__global__ void ang_triplet_kernel(const float* __restrict__ ang,
                                   const int* __restrict__ labels,
                                   float* loss_sum, unsigned int* count) {
    __shared__ float aRow[B_SIZE];
    __shared__ unsigned char flag[B_SIZE];   // 0=self, 1=pos, 2=neg
    __shared__ float fred[256];
    __shared__ int   pred[256];
    __shared__ int   nred[256];

    const int i = blockIdx.x;
    const int t = threadIdx.x;
    const int li = labels[i];

    int myPos = 0, myNeg = 0;
    for (int j = t; j < B_SIZE; j += 256) {
        const float a = ang[i * B_SIZE + j];
        const int lj = labels[j];
        unsigned char f = 0;
        if (lj == li) {
            if (j != i) { f = 1; ++myPos; }
        } else {
            f = 2; ++myNeg;
        }
        aRow[j] = a;
        flag[j] = f;
    }
    __syncthreads();

    float acc = 0.0f;
    for (int p = 0; p < B_SIZE; ++p) {
        if (flag[p] != 1) continue;
        const float ap = aRow[p] + ALPHA_RAD;
        for (int j = t; j < B_SIZE; j += 256) {
            if (flag[j] == 2) acc += fmaxf(ap - aRow[j], 0.0f);
        }
    }

    fred[t] = acc; pred[t] = myPos; nred[t] = myNeg;
    __syncthreads();
    for (int s = 128; s > 0; s >>= 1) {
        if (t < s) {
            fred[t] += fred[t + s];
            pred[t] += pred[t + s];
            nred[t] += nred[t + s];
        }
        __syncthreads();
    }
    if (t == 0) {
        atomicAdd(loss_sum, fred[0]);
        atomicAdd(count, (unsigned int)(pred[0] * nred[0]));
    }
}

// ---------------------------------------------------------------------------
// Kernel 4: final scalar.
// ---------------------------------------------------------------------------
__global__ void ang_finalize_kernel(const float* loss_sum,
                                    const unsigned int* count,
                                    float* out) {
    if (threadIdx.x == 0) {
        const float s = *loss_sum;
        const unsigned int c = *count;
        out[0] = (c > 0u) ? (s / (float)c) : s;
    }
}

// ---------------------------------------------------------------------------
extern "C" void kernel_launch(void* const* d_in, const int* in_sizes, int n_in,
                              void* d_out, int out_size, void* d_ws, size_t ws_size,
                              hipStream_t stream) {
    (void)in_sizes; (void)n_in; (void)out_size; (void)ws_size;

    const float* emb    = (const float*)d_in[0];
    const int*   labels = (const int*)d_in[1];
    float*       out    = (float*)d_out;

    char* ws = (char*)d_ws;
    _Float16*     Nh   = (_Float16*)ws;                              // 512 KB
    float*        ang  = (float*)(ws + 512u * 1024u);                // 1 MB
    float*        lsum = (float*)(ws + 512u * 1024u + 1024u * 1024u);
    unsigned int* cnt  = (unsigned int*)(ws + 512u * 1024u + 1024u * 1024u + 4u);

    ang_init_kernel<<<1, 32, 0, stream>>>(lsum, cnt);
    ang_normalize_kernel<<<B_SIZE, 256, 0, stream>>>(emb, Nh);
    // 1024 tiles of 16x16, 8 waves (tiles) per 256-thread block -> 128 blocks
    ang_wmma_sim_kernel<<<128, 256, 0, stream>>>(Nh, ang);
    ang_triplet_kernel<<<B_SIZE, 256, 0, stream>>>(ang, labels, lsum, cnt);
    ang_finalize_kernel<<<1, 32, 0, stream>>>(lsum, cnt, out);
}
